// Transformer_40407052320804
// MI455X (gfx1250) — matrix operand order repaired and checked by reading, not compiled, not run on hardware
//
#include <hip/hip_runtime.h>
#include <hip/hip_bf16.h>

typedef __bf16 bf16_t;
typedef __attribute__((ext_vector_type(16))) __bf16 v16bf;
typedef __attribute__((ext_vector_type(8)))  __bf16 v8bf;
typedef __attribute__((ext_vector_type(8)))  float  v8f;
typedef __attribute__((ext_vector_type(4)))  unsigned v4u;
typedef __attribute__((ext_vector_type(8)))  unsigned v8u;

#define D_  1024
#define H_  16
#define HS_ 64
#define L_  6
#define T_  1024
#define B_  4
#define M_  4096      // B*T
#define D4_ 4096
#define V_  50257
#define VP_ 50304     // V padded to multiple of 64

// Build the v16bf A-fragment for one lane: two contiguous 16B chunks at
// p (k+half*16) and p+8 elements (k+half*16+8), matching B's contiguous k order.
__device__ __forceinline__ v16bf load_a_frag(const bf16_t* p) {
    v8bf lo = *(const v8bf*)p;
    v8bf hi = *(const v8bf*)(p + 8);
    return __builtin_shufflevector(lo, hi,
               0,1,2,3,4,5,6,7,8,9,10,11,12,13,14,15);
}

// ---------------------------------------------------------------------------
// WMMA GEMM: C[M,N](f32) / Cbf[M,N](bf16) = alpha * A[M,K](bf16) @ B[K,N]
// B supplied TRANSPOSED: Bt[n][k] (row-major N x K, bf16) so each lane reads
// 32 contiguous bytes. Optional bias[N], residual add, ReLU. Batched via z.
// Requires: M % 128 == 0, N % 64 == 0, K % 32 == 0, lda/ldb % 16 == 0.
// 8 waves/block in 4(M) x 2(N); each wave owns a 32x32 tile (2x2 WMMAs),
// so A/B fragments are each reused twice -> 2 b128 loads per WMMA.
// ---------------------------------------------------------------------------
__global__ __launch_bounds__(256) void k_wmma_gemm_bf16(
    const bf16_t* __restrict__ A, const bf16_t* __restrict__ Bt,
    const float* __restrict__ bias, const float* __restrict__ resid,
    float* __restrict__ C, bf16_t* __restrict__ Cbf,
    int N, int K, int lda, int ldb, int ldc, int Nreal,
    float alpha, int relu,
    long long sA, long long sB, long long sC)
{
    const int tid  = threadIdx.x;
    const int lane = tid & 31;
    const int wid  = tid >> 5;
    const int wm   = wid >> 1;        // 0..3
    const int wn   = wid & 1;         // 0..1
    const int m0   = blockIdx.y * 128 + wm * 32;
    const int n0   = blockIdx.x * 64  + wn * 32;
    const int z    = blockIdx.z;

    A  += (long long)z * sA;
    Bt += (long long)z * sB;
    const long long co = (long long)z * sC;

    const int r16  = lane & 15;
    const int half = lane >> 4;

    const bf16_t* a0 = A  + (long long)(m0 + r16)      * lda + half * 16;
    const bf16_t* a1 = A  + (long long)(m0 + 16 + r16) * lda + half * 16;
    const bf16_t* b0 = Bt + (long long)(n0 + r16)      * ldb + half * 16;
    const bf16_t* b1 = Bt + (long long)(n0 + 16 + r16) * ldb + half * 16;

    v8f acc[2][2];
#pragma unroll
    for (int i = 0; i < 2; ++i)
#pragma unroll
        for (int j = 0; j < 2; ++j)
            acc[i][j] = (v8f){0.f,0.f,0.f,0.f,0.f,0.f,0.f,0.f};

    for (int k0 = 0; k0 < K; k0 += 32) {
        v16bf av0 = load_a_frag(a0 + k0);
        v16bf av1 = load_a_frag(a1 + k0);
        v16bf bv0 = *(const v16bf*)(b0 + k0);
        v16bf bv1 = *(const v16bf*)(b1 + k0);
        acc[0][0] = __builtin_amdgcn_wmma_f32_16x16x32_bf16(
                        false, av0, false, bv0, (short)0, acc[0][0], false, false);
        acc[0][1] = __builtin_amdgcn_wmma_f32_16x16x32_bf16(
                        false, av0, false, bv1, (short)0, acc[0][1], false, false);
        acc[1][0] = __builtin_amdgcn_wmma_f32_16x16x32_bf16(
                        false, av1, false, bv0, (short)0, acc[1][0], false, false);
        acc[1][1] = __builtin_amdgcn_wmma_f32_16x16x32_bf16(
                        false, av1, false, bv1, (short)0, acc[1][1], false, false);
    }

#pragma unroll
    for (int j = 0; j < 2; ++j) {
        const int  col = n0 + j * 16 + r16;
        const bool cok = col < Nreal;
        const float bval = (bias && cok) ? bias[col] : 0.0f;
#pragma unroll
        for (int i = 0; i < 2; ++i) {
#pragma unroll
            for (int r = 0; r < 8; ++r) {
                const int rm = m0 + i * 16 + half * 8 + r;
                float v = acc[i][j][r] * alpha + bval;
                const long long off = co + (long long)rm * ldc + col;
                if (resid) v += resid[off];
                if (relu)  v = fmaxf(v, 0.0f);
                if (cok) {
                    if (C)   C[off]   = v;
                    if (Cbf) Cbf[off] = (bf16_t)v;
                }
            }
        }
    }
}

// ---------------------------------------------------------------------------
// TDM L2-warm: each workgroup DMAs one 32KB chunk of a bf16 panel into LDS
// via the Tensor Data Mover, pulling it through L2 ahead of the consumer GEMM.
// D# group0/group1 per CDNA5 ISA 8.3/8.4; 2D tile 16384 x 1, data_size=2B.
// ---------------------------------------------------------------------------
#define TDM_CHUNK_BYTES 32768
#define TDM_CHUNK_ELTS  16384
__global__ __launch_bounds__(32) void k_tdm_l2warm(const bf16_t* __restrict__ src,
                                                   long long nchunks)
{
    __shared__ bf16_t stage[TDM_CHUNK_ELTS];
    if ((long long)blockIdx.x >= nchunks) return;

    const unsigned lds_off = (unsigned)(size_t)&stage[0];   // low 32b = LDS offset
    const unsigned long long ga =
        (unsigned long long)(size_t)src + (unsigned long long)blockIdx.x * TDM_CHUNK_BYTES;

    v4u g0;
    g0[0] = 1u;                                             // count=1, no gather
    g0[1] = lds_off;                                        // lds_addr
    g0[2] = (unsigned)ga;                                   // global_addr[31:0]
    g0[3] = (unsigned)((ga >> 32) & 0x01FFFFFFu)            // global_addr[56:32]
          | 0x80000000u;                                    // type=2 ("image")

    const unsigned td0 = TDM_CHUNK_ELTS, td1 = 1;           // tensor dims
    const unsigned tl0 = TDM_CHUNK_ELTS, tl1 = 1;           // tile dims
    const unsigned long long st0 = TDM_CHUNK_ELTS;          // dim0 stride

    v8u g1;
    g1[0] = (1u << 16);                                     // data_size=1 -> 2 bytes
    g1[1] = (td0 & 0xFFFFu) << 16;                          // tensor_dim0[15:0]
    g1[2] = (td0 >> 16) | ((td1 & 0xFFFFu) << 16);          // td0[31:16] | td1[15:0]
    g1[3] = (td1 >> 16) | (tl0 << 16);                      // td1[31:16] | tile_dim0
    g1[4] = tl1;                                            // tile_dim1 (tile_dim2=0)
    g1[5] = (unsigned)st0;                                  // stride0[31:0]
    g1[6] = (unsigned)(st0 >> 32) & 0xFFFFu;                // stride0[47:32]
    g1[7] = 0u;                                             // stride1 unused

    asm volatile("tensor_load_to_lds %0, %1" :: "s"(g0), "s"(g1) : "memory");
    __builtin_amdgcn_s_wait_tensorcnt(0);
}

// ---------------------------------------------------------------------------
// Weight packing: fused QKV -> bf16 [L][3D][D] with Bt[n][k] layout
// ---------------------------------------------------------------------------
__global__ void k_pack_qkv(const float* __restrict__ Wq,
                           const float* __restrict__ Wk,
                           const float* __restrict__ Wv,
                           bf16_t* __restrict__ out, long long total)
{
    for (long long i = (long long)blockIdx.x * blockDim.x + threadIdx.x;
         i < total; i += (long long)gridDim.x * blockDim.x) {
        int k = (int)(i % D_);
        long long r = i / D_;
        int n = (int)(r % (3 * D_));
        int l = (int)(r / (3 * D_));
        int sel = n >> 10;
        int nn  = n & 1023;
        int h = nn >> 6, e = nn & 63;
        const float* W = (sel == 0) ? Wq : (sel == 1) ? Wk : Wv;
        float v = W[(((long long)l * H_ + h) * D_ + k) * HS_ + e];
        out[i] = (bf16_t)v;
    }
}

// Generic transpose + fp32->bf16: out[n*K + k] = in[k*ldin + n] (0 if n>=Nreal)
__global__ void k_pack_T(const float* __restrict__ in, bf16_t* __restrict__ out,
                         int K, int ldin, int Nreal, long long total)
{
    for (long long i = (long long)blockIdx.x * blockDim.x + threadIdx.x;
         i < total; i += (long long)gridDim.x * blockDim.x) {
        int k = (int)(i % K);
        long long n = i / K;
        if (n < Nreal) {
            const float* p = in + (long long)k * ldin + n;
            __builtin_prefetch(p + 4 * (long long)ldin, 0, 1);   // global_prefetch_b8
            out[i] = (bf16_t)(*p);
        } else {
            out[i] = (bf16_t)0.0f;
        }
    }
}

__global__ void k_pad_bias(const float* __restrict__ in, float* __restrict__ out,
                           int nreal, int npad)
{
    int i = blockIdx.x * blockDim.x + threadIdx.x;
    if (i < npad) out[i] = (i < nreal) ? in[i] : 0.0f;
}

// ---------------------------------------------------------------------------
// x[row][d] = tok_emb[idx[row]][d] + pos_emb[row % T][d]
// ---------------------------------------------------------------------------
__global__ __launch_bounds__(256) void k_embed(const int* __restrict__ idx,
                                               const float* __restrict__ tok,
                                               const float* __restrict__ pos,
                                               float* __restrict__ x)
{
    const int row = blockIdx.x;
    const int t   = row & (T_ - 1);
    const long long tr = (long long)idx[row] * D_;
    for (int d = threadIdx.x; d < D_; d += 256)
        x[(long long)row * D_ + d] = tok[tr + d] + pos[(long long)t * D_ + d];
}

// ---------------------------------------------------------------------------
// In-place LayerNorm over D=1024, optional bf16 mirror for WMMA A-operand.
// ---------------------------------------------------------------------------
__global__ __launch_bounds__(256) void k_layernorm(float* __restrict__ x,
                                                   const float* __restrict__ s,
                                                   const float* __restrict__ b,
                                                   bf16_t* __restrict__ xbf)
{
    __shared__ float red[256];
    const int row = blockIdx.x, tid = threadIdx.x;
    float* xr = x + (long long)row * D_;

    float acc = 0.0f;
    for (int d = tid; d < D_; d += 256) acc += xr[d];
    red[tid] = acc; __syncthreads();
    for (int st = 128; st > 0; st >>= 1) {
        if (tid < st) red[tid] += red[tid + st];
        __syncthreads();
    }
    const float mean = red[0] * (1.0f / D_);
    __syncthreads();

    float vacc = 0.0f;
    for (int d = tid; d < D_; d += 256) { float t = xr[d] - mean; vacc += t * t; }
    red[tid] = vacc; __syncthreads();
    for (int st = 128; st > 0; st >>= 1) {
        if (tid < st) red[tid] += red[tid + st];
        __syncthreads();
    }
    const float rstd = rsqrtf(red[0] * (1.0f / D_) + 1e-5f);

    for (int d = tid; d < D_; d += 256) {
        float y = (xr[d] - mean) * rstd * s[d] + b[d];
        xr[d] = y;
        if (xbf) xbf[(long long)row * D_ + d] = (bf16_t)y;
    }
}

// ---------------------------------------------------------------------------
// qkv[M,3D] bf16 -> qb,kb [B,H,T,HS] and v transposed vTb [B,H,HS,T]
// ---------------------------------------------------------------------------
__global__ void k_split_qkv(const bf16_t* __restrict__ qkv,
                            bf16_t* __restrict__ qb,
                            bf16_t* __restrict__ kb,
                            bf16_t* __restrict__ vTb)
{
    long long i = (long long)blockIdx.x * blockDim.x + threadIdx.x;
    if (i >= (long long)M_ * D_) return;
    const int d = (int)(i & (D_ - 1));
    const long long row = i >> 10;
    const int t = (int)(row & (T_ - 1));
    const int b = (int)(row >> 10);
    const int h = d >> 6, e = d & 63;
    const long long base = row * (3 * D_);
    const long long zo = (long long)(b * H_ + h);
    qb [(zo * T_  + t) * HS_ + e] = qkv[base + d];
    kb [(zo * T_  + t) * HS_ + e] = qkv[base + D_ + d];
    vTb[(zo * HS_ + e) * T_  + t] = qkv[base + 2 * D_ + d];
}

// ---------------------------------------------------------------------------
// Reference softmaxes over the QUERY axis t with causal mask (valid: t >= s).
// S[s,t] = K·Qᵀ (softmax is over the row of S). Writes transposed bf16 P[t][s].
// ---------------------------------------------------------------------------
__global__ __launch_bounds__(256) void k_softmax_colT(const float* __restrict__ S,
                                                      bf16_t* __restrict__ P)
{
    __shared__ float red[256];
    const int srow = blockIdx.x;
    const int h    = blockIdx.y;
    const int tid  = threadIdx.x;
    const float* Sr = S + (long long)h * T_ * T_ + (long long)srow * T_;
    bf16_t*      Pb = P + (long long)h * T_ * T_ + srow;

    float m = -3.4e38f;
    for (int t = srow + tid; t < T_; t += 256) m = fmaxf(m, Sr[t]);
    red[tid] = m; __syncthreads();
    for (int st = 128; st > 0; st >>= 1) {
        if (tid < st) red[tid] = fmaxf(red[tid], red[tid + st]);
        __syncthreads();
    }
    m = red[0]; __syncthreads();

    float sum = 0.0f;
    for (int t = srow + tid; t < T_; t += 256) sum += __expf(Sr[t] - m);
    red[tid] = sum; __syncthreads();
    for (int st = 128; st > 0; st >>= 1) {
        if (tid < st) red[tid] += red[tid + st];
        __syncthreads();
    }
    const float inv = 1.0f / red[0];

    for (int t = tid; t < T_; t += 256) {
        float p = (t >= srow) ? __expf(Sr[t] - m) * inv : 0.0f;
        Pb[(long long)t * T_] = (bf16_t)p;
    }
}

__global__ void k_zero1(float* p) { *p = 0.0f; }

// ---------------------------------------------------------------------------
// loss += (logsumexp(logits[row]) - logits[row][tgt]) / M   (one block/row)
// ---------------------------------------------------------------------------
__global__ __launch_bounds__(256) void k_loss(const float* __restrict__ logits,
                                              const int* __restrict__ tgt,
                                              float* __restrict__ loss)
{
    __shared__ float red[256];
    const int row = blockIdx.x, tid = threadIdx.x;
    const float* lr = logits + (long long)row * V_;

    float m = -3.4e38f;
    for (int j = tid; j < V_; j += 256) m = fmaxf(m, lr[j]);
    red[tid] = m; __syncthreads();
    for (int st = 128; st > 0; st >>= 1) {
        if (tid < st) red[tid] = fmaxf(red[tid], red[tid + st]);
        __syncthreads();
    }
    m = red[0]; __syncthreads();

    float s = 0.0f;
    for (int j = tid; j < V_; j += 256) s += __expf(lr[j] - m);
    red[tid] = s; __syncthreads();
    for (int st = 128; st > 0; st >>= 1) {
        if (tid < st) red[tid] += red[tid + st];
        __syncthreads();
    }
    if (tid == 0) {
        const float lse = m + __logf(red[0]);
        atomicAdd(loss, (lse - lr[tgt[row]]) * (1.0f / (float)M_));
    }
}

// ---------------------------------------------------------------------------
extern "C" void kernel_launch(void* const* d_in, const int* in_sizes, int n_in,
                              void* d_out, int out_size, void* d_ws, size_t ws_size,
                              hipStream_t stream)
{
    (void)in_sizes; (void)n_in; (void)out_size; (void)ws_size;

    const int*   idx    = (const int*)  d_in[0];
    const int*   target = (const int*)  d_in[1];
    const float* tok    = (const float*)d_in[2];
    const float* pos    = (const float*)d_in[3];
    const float* Wq     = (const float*)d_in[4];
    const float* Wk     = (const float*)d_in[5];
    const float* Wv     = (const float*)d_in[6];
    const float* Wo     = (const float*)d_in[7];
    const float* bo     = (const float*)d_in[8];
    const float* ln1s   = (const float*)d_in[9];
    const float* ln1b   = (const float*)d_in[10];
    const float* ln2s   = (const float*)d_in[11];
    const float* ln2b   = (const float*)d_in[12];
    const float* W1     = (const float*)d_in[13];
    const float* b1     = (const float*)d_in[14];
    const float* W2     = (const float*)d_in[15];
    const float* b2     = (const float*)d_in[16];
    const float* lnfs   = (const float*)d_in[17];
    const float* lnfb   = (const float*)d_in[18];
    const float* Wlm    = (const float*)d_in[19];
    const float* blm    = (const float*)d_in[20];

    float* logits = (float*)d_out;
    float* loss   = logits + (size_t)M_ * V_;

    // ---- carve workspace ----
    char* wsp = (char*)d_ws;
    auto carve = [&](size_t bytes) -> void* {
        void* p = (void*)wsp;
        wsp += (bytes + 255) & ~(size_t)255;
        return p;
    };
    bf16_t* wqkvT  = (bf16_t*)carve((size_t)L_ * 3 * D_ * D_ * 2);   // [L][3D][D]
    bf16_t* woT    = (bf16_t*)carve((size_t)L_ * D_ * D_ * 2);       // [L][D][D]
    bf16_t* w1T    = (bf16_t*)carve((size_t)L_ * D4_ * D_ * 2);      // [L][4D][D]
    bf16_t* w2T    = (bf16_t*)carve((size_t)L_ * D_ * D4_ * 2);      // [L][D][4D]
    bf16_t* wlmT   = (bf16_t*)carve((size_t)VP_ * D_ * 2);           // [VP][D]
    float*  blmP   = (float*) carve((size_t)VP_ * 4);
    float*  x      = (float*) carve((size_t)M_ * D_ * 4);            // residual stream
    bf16_t* xbf    = (bf16_t*)carve((size_t)M_ * D_ * 2);
    bf16_t* hbf    = (bf16_t*)carve((size_t)M_ * D4_ * 2);
    bf16_t* qkvbf  = (bf16_t*)carve((size_t)M_ * 3 * D_ * 2);
    bf16_t* qb     = (bf16_t*)carve((size_t)B_ * H_ * T_ * HS_ * 2);
    bf16_t* kb     = (bf16_t*)carve((size_t)B_ * H_ * T_ * HS_ * 2);
    bf16_t* vTb    = (bf16_t*)carve((size_t)B_ * H_ * HS_ * T_ * 2);
    float*  S      = (float*) carve((size_t)H_ * T_ * T_ * 4);       // per-batch reuse
    bf16_t* P      = (bf16_t*)carve((size_t)H_ * T_ * T_ * 2);       // per-batch reuse
    bf16_t* attcat = (bf16_t*)carve((size_t)M_ * D_ * 2);

    auto gemm = [&](const bf16_t* A, const bf16_t* Bt, const float* bias,
                    const float* resid, float* C, bf16_t* Cbf,
                    int Mloc, int N, int K, int lda, int ldb, int ldc,
                    int Nreal, float alpha, int relu,
                    long long sA, long long sB, long long sC, int batch) {
        dim3 g(N / 64, Mloc / 128, batch);
        k_wmma_gemm_bf16<<<g, 256, 0, stream>>>(A, Bt, bias, resid, C, Cbf,
                                                N, K, lda, ldb, ldc, Nreal,
                                                alpha, relu, sA, sB, sC);
    };

    // ---- pack all weights to bf16 (transposed) once ----
    k_pack_qkv<<<4096, 256, 0, stream>>>(Wq, Wk, Wv, wqkvT, (long long)L_ * 3 * D_ * D_);
    for (int l = 0; l < L_; ++l) {
        k_pack_T<<<2048, 256, 0, stream>>>(Wo + (size_t)l * D_ * D_,
                                           woT + (size_t)l * D_ * D_,
                                           D_, D_, D_, (long long)D_ * D_);
        k_pack_T<<<4096, 256, 0, stream>>>(W1 + (size_t)l * D_ * D4_,
                                           w1T + (size_t)l * D4_ * D_,
                                           D_, D4_, D4_, (long long)D4_ * D_);
        k_pack_T<<<4096, 256, 0, stream>>>(W2 + (size_t)l * D4_ * D_,
                                           w2T + (size_t)l * D_ * D4_,
                                           D4_, D_, D_, (long long)D_ * D4_);
    }
    k_pack_T<<<8192, 256, 0, stream>>>(Wlm, wlmT, D_, V_, V_, (long long)VP_ * D_);
    k_pad_bias<<<(VP_ + 255) / 256, 256, 0, stream>>>(blm, blmP, V_, VP_);

    // ---- embedding ----
    k_embed<<<M_, 256, 0, stream>>>(idx, tok, pos, x);

    const float scale = 0.03125f; // D^-0.5 = 1/32 (reference uses NUM_EMBED)

    // ---- transformer layers ----
    for (int l = 0; l < L_; ++l) {
        // x = LN1(x); xbf = bf16(x)
        k_layernorm<<<M_, 256, 0, stream>>>(x, ln1s + l * D_, ln1b + l * D_, xbf);

        // qkv[M,3D] = x @ [Wq|Wk|Wv]
        gemm(xbf, wqkvT + (size_t)l * 3 * D_ * D_, nullptr, nullptr,
             nullptr, qkvbf, M_, 3 * D_, D_, D_, D_, 3 * D_, 3 * D_,
             1.0f, 0, 0, 0, 0, 1);

        k_split_qkv<<<(M_ * D_) / 256, 256, 0, stream>>>(qkvbf, qb, kb, vTb);

        for (int b = 0; b < B_; ++b) {
            const long long bo_ = (long long)b * H_ * T_ * HS_;
            // S[s,t] = scale * K[s,:]·Q[t,:]   (batched over heads)
            gemm(kb + bo_, qb + bo_, nullptr, nullptr, S, nullptr,
                 T_, T_, HS_, HS_, HS_, T_, T_, scale, 0,
                 (long long)T_ * HS_, (long long)T_ * HS_, (long long)T_ * T_, H_);
            // masked softmax over query axis; write Pᵀ[t][s] bf16
            k_softmax_colT<<<dim3(T_, H_), 256, 0, stream>>>(S, P);
            // att[t,e] = Σ_s P[t,s]·V[s,e]  -> written into head-concat layout
            gemm(P, vTb + (long long)b * H_ * HS_ * T_, nullptr, nullptr,
                 nullptr, attcat + (long long)b * T_ * D_,
                 T_, HS_, T_, T_, T_, D_, HS_, 1.0f, 0,
                 (long long)T_ * T_, (long long)HS_ * T_, 64LL, H_);
        }

        // x = x + attcat @ Wo + bo
        gemm(attcat, woT + (size_t)l * D_ * D_, bo + l * D_, x, x, nullptr,
             M_, D_, D_, D_, D_, D_, D_, 1.0f, 0, 0, 0, 0, 1);

        // x = LN2(x); xbf = bf16(x)
        k_layernorm<<<M_, 256, 0, stream>>>(x, ln2s + l * D_, ln2b + l * D_, xbf);

        // h = relu(x @ W1 + b1)  (bf16 out only)
        gemm(xbf, w1T + (size_t)l * D4_ * D_, b1 + l * D4_, nullptr,
             nullptr, hbf, M_, D4_, D_, D_, D_, D4_, D4_,
             1.0f, 1, 0, 0, 0, 1);

        // x = x + h @ W2 + b2
        gemm(hbf, w2T + (size_t)l * D_ * D4_, b2 + l * D_, x, x, nullptr,
             M_, D_, D4_, D4_, D4_, D_, D_, 1.0f, 0, 0, 0, 0, 1);
    }

    // ---- final LN + logits + loss ----
    k_layernorm<<<M_, 256, 0, stream>>>(x, lnfs, lnfb, xbf);

    // TDM warm: stream the logits weight panel (103MB bf16, L2-resident) through L2
    {
        const long long nchunks = ((long long)VP_ * D_ * 2) / TDM_CHUNK_BYTES;
        k_tdm_l2warm<<<(unsigned)nchunks, 32, 0, stream>>>(wlmT, nchunks);
    }

    gemm(xbf, wlmT, blmP, nullptr, logits, nullptr,
         M_, VP_, D_, D_, D_, V_, V_, 1.0f, 0, 0, 0, 0, 1);

    k_zero1<<<1, 1, 0, stream>>>(loss);
    k_loss<<<M_, 256, 0, stream>>>(logits, target, loss);
}
